// Similarity_55980603736130
// MI455X (gfx1250) — matrix operand order repaired and checked by reading, not compiled, not run on hardware
//
#include <hip/hip_runtime.h>
#include <hip/hip_bf16.h>

// Shapes fixed by the reference
#define BB 32
#define AA 2048
#define TT 512
#define DD 256
#define TOPK 10

typedef __attribute__((ext_vector_type(16))) __bf16 v16bf;
typedef __attribute__((ext_vector_type(8)))  float  v8f;
typedef unsigned int u32;
typedef __attribute__((ext_vector_type(4))) u32 u32x4;
typedef __attribute__((ext_vector_type(8))) int i32x8;
typedef __attribute__((ext_vector_type(4))) int i32x4;

#if __has_builtin(__builtin_amdgcn_tensor_load_to_lds) && __has_builtin(__builtin_amdgcn_s_wait_tensorcnt)
#define USE_TDM 1
#else
#define USE_TDM 0
#endif

union ABfrag {
    unsigned int u[8];
    v16bf        v;
};

// f32 -> bf16 round-to-nearest-even (NaN edge ignored; data is finite)
static __device__ __forceinline__ unsigned int f2bf(float f) {
    unsigned int u = __float_as_uint(f);
    u += 0x7FFFu + ((u >> 16) & 1u);
    return u >> 16;
}
static __device__ __forceinline__ unsigned int pk(float lo, float hi) {
    return f2bf(lo) | (f2bf(hi) << 16);
}
// low 32 bits of a generic pointer to LDS = byte offset within the
// workgroup's LDS allocation (ISA 10.2).
static __device__ __forceinline__ unsigned int lds_off(const void* p) {
    return (unsigned int)(unsigned long long)p;
}

#if USE_TDM
// Issue a TDM 2D tile load: 128 rows x 32 bf16, row stride 256 elems in
// global, LDS rows padded by 16B after each 64B row (-> 40-elem stride).
// This toolchain declares the 6-arg builtin:
//   (uint32x4 g0, int32x8 g1, int32x4 g2, int32x4 g3, int32x8 pad, i32 cpol)
static __device__ __forceinline__ void tdm_load_slab(unsigned long long gaddr,
                                                     unsigned int ldsOff) {
    u32x4 g0;
    g0[0] = 1u;                                             // count=1, load, user
    g0[1] = ldsOff;                                         // lds_addr
    g0[2] = (u32)(gaddr & 0xFFFFFFFFull);                   // global_addr lo
    g0[3] = (u32)((gaddr >> 32) & 0x1FFFFFFull) | (2u << 30); // addr hi | type=2
    i32x8 g1;
    g1[0] = (int)((1u << 16)      // data_size = 2 bytes
                | (1u << 20)      // pad_enable
                | (3u << 22)      // pad_interval: every 16 DWORDs (64B row)
                | (3u << 25));    // pad_amount: 4 DWORDs (16B)
    g1[1] = (int)(256u << 16);    // tensor_dim0 = 256 (low 16)
    g1[2] = (int)(128u << 16);    // tensor_dim0 hi = 0 | tensor_dim1 = 128
    g1[3] = (int)(32u << 16);     // tensor_dim1 hi = 0 | tile_dim0 = 32
    g1[4] = (int)(128u);          // tile_dim1 = 128 | tile_dim2 = 0
    g1[5] = (int)(256u);          // tensor_dim0_stride = 256 (lo)
    g1[6] = 0;                    // stride hi | tensor_dim1_stride lo
    g1[7] = 0;
    i32x4 z4 = {0, 0, 0, 0};
    i32x8 z8 = {0, 0, 0, 0, 0, 0, 0, 0};
    __builtin_amdgcn_tensor_load_to_lds(g0, g1, z4, z4, z8, 0);
}
#endif

// ---------------------------------------------------------------------------
// Kernel 0 (fast path): fused f32->bf16 conversion + squared L2 norms.
// One wave32 per row; streams 80MB in, 40MB bf16 out (HBM-bound, VALU free).
// ---------------------------------------------------------------------------
__global__ __launch_bounds__(256) void convert_sqnorm_kernel(
    const float* __restrict__ article, const float* __restrict__ title,
    unsigned short* __restrict__ abf, unsigned short* __restrict__ tbf,
    float* __restrict__ asq, float* __restrict__ tsq)
{
    const int NA = BB * AA;
    const int NT = BB * TT;
    int wid  = (blockIdx.x * blockDim.x + threadIdx.x) >> 5;
    int lane = threadIdx.x & 31;
    if (wid >= NA + NT) return;
    const float* src = (wid < NA) ? (article + (size_t)wid * DD)
                                  : (title + (size_t)(wid - NA) * DD);
    unsigned short* dst = (wid < NA) ? (abf + (size_t)wid * DD)
                                     : (tbf + (size_t)(wid - NA) * DD);
    const float4* p = (const float4*)src + lane * 2;   // 8 floats per lane
    float4 x = p[0], y = p[1];
    uint4 o;
    o.x = pk(x.x, x.y); o.y = pk(x.z, x.w);
    o.z = pk(y.x, y.y); o.w = pk(y.z, y.w);
    ((uint4*)dst)[lane] = o;
    float s = x.x*x.x + x.y*x.y + x.z*x.z + x.w*x.w
            + y.x*y.x + y.y*y.y + y.z*y.z + y.w*y.w;
    #pragma unroll
    for (int m = 16; m >= 1; m >>= 1) s += __shfl_xor(s, m, 32);
    if (lane == 0) {
        if (wid < NA) asq[wid] = s; else tsq[wid - NA] = s;
    }
}

// Kernel 0 (fallback path): norms only.
__global__ __launch_bounds__(256) void sqnorm_kernel(
    const float* __restrict__ article, const float* __restrict__ title,
    float* __restrict__ asq, float* __restrict__ tsq)
{
    const int NA = BB * AA;
    const int NT = BB * TT;
    int wid  = (blockIdx.x * blockDim.x + threadIdx.x) >> 5;
    int lane = threadIdx.x & 31;
    if (wid >= NA + NT) return;
    const float* src = (wid < NA) ? (article + (size_t)wid * DD)
                                  : (title + (size_t)(wid - NA) * DD);
    const float4* p = (const float4*)src + lane * 2;
    float4 x = p[0], y = p[1];
    float s = x.x*x.x + x.y*x.y + x.z*x.z + x.w*x.w
            + y.x*y.x + y.y*y.y + y.z*y.z + y.w*y.w;
    #pragma unroll
    for (int m = 16; m >= 1; m >>= 1) s += __shfl_xor(s, m, 32);
    if (lane == 0) {
        if (wid < NA) asq[wid] = s; else tsq[wid - NA] = s;
    }
}

// ---------------------------------------------------------------------------
// Kernel 1 (fast path): bf16 WMMA GEMM from pre-converted bf16 tensors.
// Block: 256 threads (8 waves). Tile: 128 A-rows x 128 T-cols, K-step 32.
// LDS slabs double-buffered; filled by the Tensor Data Mover (wave 0 issues,
// s_wait_tensorcnt + workgroup barrier publish) when available.
// Fused epilogue: s = tmask * exp(2c - |a|^2 - |t|^2), row-max over T.
// ---------------------------------------------------------------------------
__global__ __launch_bounds__(256) void gemm_rowmax_bf16_kernel(
    const unsigned short* __restrict__ abf, const unsigned short* __restrict__ tbf,
    const float* __restrict__ amask,   const float* __restrict__ tmask,
    const float* __restrict__ asq_g,   const float* __restrict__ tsq_g,
    float* __restrict__ rowmax_g)
{
    // 40-elem (20-uint) padded row stride, double buffered
    __shared__ unsigned int AslU[2][128 * 20];
    __shared__ unsigned int TslU[2][128 * 20];

    const int b  = blockIdx.y;
    const int a0 = blockIdx.x * 128;

    const int lane = threadIdx.x & 31;
    const int w    = threadIdx.x >> 5;
    const int n0   = lane & 15;
    const int half = lane >> 4;

    float asqr[8];
    #pragma unroll
    for (int r = 0; r < 8; ++r)
        asqr[r] = asq_g[b * AA + a0 + w * 16 + half * 8 + r];

    float rmax8[8];
    #pragma unroll
    for (int r = 0; r < 8; ++r) rmax8[r] = 0.0f;

#if USE_TDM
    const unsigned long long aBase = (unsigned long long)abf +
        ((unsigned long long)(b * AA + a0)) * DD * 2ull;
    const unsigned long long tBase = (unsigned long long)tbf +
        ((unsigned long long)(b * TT)) * DD * 2ull;
    // stage(step g): A slab at k0, B slab at (tc*128 rows, k0)
    #define STAGE(buf, tc_, k0_)                                              \
        do { if (w == 0) {                                                    \
            tdm_load_slab(aBase + (unsigned long long)(k0_) * 2ull,           \
                          lds_off(&AslU[(buf)][0]));                          \
            tdm_load_slab(tBase + ((unsigned long long)(tc_) * 128ull * DD +  \
                                   (unsigned long long)(k0_)) * 2ull,         \
                          lds_off(&TslU[(buf)][0]));                          \
        } } while (0)
#else
    const int srow = threadIdx.x >> 1;
    const int sseg = threadIdx.x & 1;
    #define STAGE(buf, tc_, k0_)                                              \
        do {                                                                  \
            const uint4* ap = (const uint4*)(abf +                            \
                ((size_t)(b * AA + a0 + srow)) * DD + (k0_) + sseg * 16);     \
            uint4 av0 = ap[0], av1 = ap[1];                                   \
            uint4* ad = (uint4*)&AslU[(buf)][srow * 20 + sseg * 8];           \
            ad[0] = av0; ad[1] = av1;                                         \
            const uint4* tp = (const uint4*)(tbf +                            \
                ((size_t)(b * TT + (tc_) * 128 + srow)) * DD + (k0_) + sseg * 16); \
            uint4 tv0 = tp[0], tv1 = tp[1];                                   \
            uint4* td = (uint4*)&TslU[(buf)][srow * 20 + sseg * 8];           \
            td[0] = tv0; td[1] = tv1;                                         \
        } while (0)
#endif

    STAGE(0, 0, 0);                       // prologue: step 0 -> buffer 0

    for (int tc = 0; tc < TT / 128; ++tc) {
        v8f acc[8] = {};
        for (int kk = 0; kk < DD / 32; ++kk) {
            const int g   = tc * 8 + kk;
            const int cur = g & 1;
#if USE_TDM
            if (w == 0) __builtin_amdgcn_s_wait_tensorcnt(0);
#endif
            __syncthreads();              // buf[cur] ready; buf[1-cur] free
            if (g + 1 < 32) {
                const int g2  = g + 1;
                STAGE(g2 & 1, g2 >> 3, (g2 & 7) * 32);
            }

            const unsigned int* As = AslU[cur];
            const unsigned int* Ts = TslU[cur];

            ABfrag afr;                   // A 16x32 bf16 fragment (ISA 7.12.2)
            const int arow = w * 16 + n0;
            #pragma unroll
            for (int v = 0; v < 4; ++v) {
                afr.u[v]     = As[arow * 20 + v + 4 * half];
                afr.u[4 + v] = As[arow * 20 + 8 + v + 4 * half];
            }
            #pragma unroll
            for (int ct = 0; ct < 8; ++ct) {
                ABfrag bfr;               // B 32x16 fragment
                const int trow = ct * 16 + n0;
                #pragma unroll
                for (int j = 0; j < 8; ++j)
                    bfr.u[j] = Ts[trow * 20 + 8 * (j >> 2) + 4 * half + (j & 3)];
                acc[ct] = __builtin_amdgcn_wmma_f32_16x16x32_bf16(
                    false, afr.v, false, bfr.v, (short)0, acc[ct], false, false);
            }
        }

        // fused epilogue (registers only)
        #pragma unroll
        for (int ct = 0; ct < 8; ++ct) {
            const int tg = b * TT + tc * 128 + ct * 16 + n0;
            const float tsq = tsq_g[tg];
            const float tm  = tmask[tg];
            #pragma unroll
            for (int r = 0; r < 8; ++r) {
                float s = tm * __expf(2.0f * acc[ct][r] - asqr[r] - tsq);
                rmax8[r] = fmaxf(rmax8[r], s);
            }
        }
    }
    #undef STAGE

    #pragma unroll
    for (int r = 0; r < 8; ++r) {
        float v = rmax8[r];
        v = fmaxf(v, __shfl_xor(v, 1, 32));
        v = fmaxf(v, __shfl_xor(v, 2, 32));
        v = fmaxf(v, __shfl_xor(v, 4, 32));
        v = fmaxf(v, __shfl_xor(v, 8, 32));
        if (n0 == 0) {
            const int ag = a0 + w * 16 + half * 8 + r;
            rowmax_g[b * AA + ag] = amask[b * AA + ag] * v;
        }
    }
}

// ---------------------------------------------------------------------------
// Kernel 1 (fallback, small workspace): convert-in-kernel version (round 1).
// ---------------------------------------------------------------------------
__global__ __launch_bounds__(256) void gemm_rowmax_conv_kernel(
    const float* __restrict__ article, const float* __restrict__ title,
    const float* __restrict__ amask,   const float* __restrict__ tmask,
    const float* __restrict__ asq_g,   const float* __restrict__ tsq_g,
    float* __restrict__ rowmax_g)
{
    __shared__ unsigned int AslU[128 * 20];
    __shared__ unsigned int TslU[128 * 20];

    const int b  = blockIdx.y;
    const int a0 = blockIdx.x * 128;
    const int lane = threadIdx.x & 31;
    const int w    = threadIdx.x >> 5;
    const int n0   = lane & 15;
    const int half = lane >> 4;

    float asqr[8];
    #pragma unroll
    for (int r = 0; r < 8; ++r)
        asqr[r] = asq_g[b * AA + a0 + w * 16 + half * 8 + r];

    float rmax8[8];
    #pragma unroll
    for (int r = 0; r < 8; ++r) rmax8[r] = 0.0f;

    const int srow = threadIdx.x >> 1;
    const int sseg = threadIdx.x & 1;

    for (int tc = 0; tc < TT / 128; ++tc) {
        v8f acc[8] = {};
        for (int k0 = 0; k0 < DD; k0 += 32) {
            __syncthreads();
            {
                const float4* ap = (const float4*)(article +
                    ((size_t)(b * AA + a0 + srow)) * DD + k0 + sseg * 16);
                float4 x0 = ap[0], x1 = ap[1], x2 = ap[2], x3 = ap[3];
                unsigned int* d = &AslU[srow * 20 + sseg * 8];
                d[0] = pk(x0.x, x0.y); d[1] = pk(x0.z, x0.w);
                d[2] = pk(x1.x, x1.y); d[3] = pk(x1.z, x1.w);
                d[4] = pk(x2.x, x2.y); d[5] = pk(x2.z, x2.w);
                d[6] = pk(x3.x, x3.y); d[7] = pk(x3.z, x3.w);
                const float4* tp = (const float4*)(title +
                    ((size_t)(b * TT + tc * 128 + srow)) * DD + k0 + sseg * 16);
                float4 y0 = tp[0], y1 = tp[1], y2 = tp[2], y3 = tp[3];
                unsigned int* e = &TslU[srow * 20 + sseg * 8];
                e[0] = pk(y0.x, y0.y); e[1] = pk(y0.z, y0.w);
                e[2] = pk(y1.x, y1.y); e[3] = pk(y1.z, y1.w);
                e[4] = pk(y2.x, y2.y); e[5] = pk(y2.z, y2.w);
                e[6] = pk(y3.x, y3.y); e[7] = pk(y3.z, y3.w);
            }
            __syncthreads();

            ABfrag afr;
            const int arow = w * 16 + n0;
            #pragma unroll
            for (int v = 0; v < 4; ++v) {
                afr.u[v]     = AslU[arow * 20 + v + 4 * half];
                afr.u[4 + v] = AslU[arow * 20 + 8 + v + 4 * half];
            }
            #pragma unroll
            for (int ct = 0; ct < 8; ++ct) {
                ABfrag bfr;
                const int trow = ct * 16 + n0;
                #pragma unroll
                for (int j = 0; j < 8; ++j)
                    bfr.u[j] = TslU[trow * 20 + 8 * (j >> 2) + 4 * half + (j & 3)];
                acc[ct] = __builtin_amdgcn_wmma_f32_16x16x32_bf16(
                    false, afr.v, false, bfr.v, (short)0, acc[ct], false, false);
            }
        }
        #pragma unroll
        for (int ct = 0; ct < 8; ++ct) {
            const int tg = b * TT + tc * 128 + ct * 16 + n0;
            const float tsq = tsq_g[tg];
            const float tm  = tmask[tg];
            #pragma unroll
            for (int r = 0; r < 8; ++r) {
                float s = tm * __expf(2.0f * acc[ct][r] - asqr[r] - tsq);
                rmax8[r] = fmaxf(rmax8[r], s);
            }
        }
    }

    #pragma unroll
    for (int r = 0; r < 8; ++r) {
        float v = rmax8[r];
        v = fmaxf(v, __shfl_xor(v, 1, 32));
        v = fmaxf(v, __shfl_xor(v, 2, 32));
        v = fmaxf(v, __shfl_xor(v, 4, 32));
        v = fmaxf(v, __shfl_xor(v, 8, 32));
        if (n0 == 0) {
            const int ag = a0 + w * 16 + half * 8 + r;
            rowmax_g[b * AA + ag] = amask[b * AA + ag] * v;
        }
    }
}

// ---------------------------------------------------------------------------
// Kernel 2: per-batch top-10 (descending, as lax.top_k) + tiny MLP.
// ---------------------------------------------------------------------------
__global__ __launch_bounds__(256) void topk_mlp_kernel(
    const float* __restrict__ rowmax,
    const float* __restrict__ W1, const float* __restrict__ b1,
    const float* __restrict__ W2, const float* __restrict__ b2,
    float* __restrict__ out)
{
    __shared__ float vals[AA];
    __shared__ float redv[256];
    __shared__ int   redi[256];
    __shared__ float top[TOPK];

    const int b = blockIdx.x;
    const int t = threadIdx.x;
    for (int i = t; i < AA; i += 256) vals[i] = rowmax[b * AA + i];
    __syncthreads();

    for (int it = 0; it < TOPK; ++it) {
        float mv = -1.0f; int mi = 0x7FFFFFFF;
        for (int i = t; i < AA; i += 256) {
            float v = vals[i];
            if (v > mv || (v == mv && i < mi)) { mv = v; mi = i; }
        }
        redv[t] = mv; redi[t] = mi;
        __syncthreads();
        for (int s = 128; s > 0; s >>= 1) {
            if (t < s) {
                if (redv[t + s] > redv[t] ||
                    (redv[t + s] == redv[t] && redi[t + s] < redi[t])) {
                    redv[t] = redv[t + s]; redi[t] = redi[t + s];
                }
            }
            __syncthreads();
        }
        if (t == 0) { top[it] = redv[0]; vals[redi[0]] = -1.0f; }
        __syncthreads();
    }

    if (t == 0) {
        float o = b2[0];
        #pragma unroll
        for (int j = 0; j < TOPK; ++j) {
            float h = b1[j];
            #pragma unroll
            for (int k = 0; k < TOPK; ++k) h += W1[j * TOPK + k] * top[k];
            o += W2[j] * fmaxf(h, 0.0f);
        }
        out[b] = o;
    }
}

// ---------------------------------------------------------------------------
extern "C" void kernel_launch(void* const* d_in, const int* in_sizes, int n_in,
                              void* d_out, int out_size, void* d_ws, size_t ws_size,
                              hipStream_t stream)
{
    (void)in_sizes; (void)n_in; (void)out_size;
    const float* article = (const float*)d_in[0];
    const float* title   = (const float*)d_in[1];
    const float* amask   = (const float*)d_in[2];
    const float* tmask   = (const float*)d_in[3];
    const float* W1      = (const float*)d_in[4];
    const float* b1      = (const float*)d_in[5];
    const float* W2      = (const float*)d_in[6];
    const float* b2      = (const float*)d_in[7];
    float* out = (float*)d_out;

    const size_t nABf = (size_t)BB * AA * DD;      // bf16 article elems
    const size_t nTBf = (size_t)BB * TT * DD;      // bf16 title elems
    const size_t needFast = (nABf + nTBf) * 2 +
                            ((size_t)BB * AA + BB * TT + BB * AA) * 4;
    const int rows = BB * AA + BB * TT;            // one wave32 per row

    if (ws_size >= needFast) {
        unsigned short* abf = (unsigned short*)d_ws;
        unsigned short* tbf = abf + nABf;
        float* asq    = (float*)(tbf + nTBf);
        float* tsq    = asq + BB * AA;
        float* rowmax = tsq + BB * TT;

        convert_sqnorm_kernel<<<rows / 8, 256, 0, stream>>>(
            article, title, abf, tbf, asq, tsq);
        gemm_rowmax_bf16_kernel<<<dim3(AA / 128, BB), 256, 0, stream>>>(
            abf, tbf, amask, tmask, asq, tsq, rowmax);
        topk_mlp_kernel<<<BB, 256, 0, stream>>>(rowmax, W1, b1, W2, b2, out);
    } else {
        float* asq    = (float*)d_ws;
        float* tsq    = asq + BB * AA;
        float* rowmax = tsq + BB * TT;

        sqnorm_kernel<<<rows / 8, 256, 0, stream>>>(article, title, asq, tsq);
        gemm_rowmax_conv_kernel<<<dim3(AA / 128, BB), 256, 0, stream>>>(
            article, title, amask, tmask, asq, tsq, rowmax);
        topk_mlp_kernel<<<BB, 256, 0, stream>>>(rowmax, W1, b1, W2, b2, out);
    }
}
